// AttentionModel_68264210203181
// MI455X (gfx1250) — matrix operand order repaired and checked by reading, not compiled, not run on hardware
//
#include <hip/hip_runtime.h>
#include <hip/hip_bf16.h>

// ---------------------------------------------------------------------------
// CDNA5 WMMA (wave32): D(16x16 f32) = A(16x32 bf16) x B(32x16 bf16) + C
// A and B tiles are kept in "fragment order": for each (tile, lane) the 16
// bf16 elements a lane owns are contiguous -> one 32-byte vector load.
// LDS B tiles are double-buffered: one barrier per K-step, staging overlaps
// the WMMA issue.  NT2=4 N-tiles per block -> 4 WMMA per wave per barrier,
// with all B fragments loaded before the WMMA chain issues.
// ---------------------------------------------------------------------------
typedef __attribute__((ext_vector_type(16))) __bf16          v16bf;
typedef __attribute__((ext_vector_type(16))) unsigned short  v16us;
typedef __attribute__((ext_vector_type(8)))  float           v8f;

#define L      256
#define PIX    65536        // 256*256
#define CH     64
#define NT2    4            // n-tiles (of 16 pixels) per block

static __device__ __forceinline__ unsigned short bfbits(float f) {
    return __builtin_bit_cast(unsigned short, __float2bfloat16(f));   // HW cvt, RNE
}

// A-fragment element e (0..15) -> K offset inside a 16x32 tile (contiguous: k = 16*(lane/16) + e)
static __device__ __forceinline__ int a_koff(int e, int lane) {
    return e + ((lane & 16) ? 16 : 0);
}
// B element (k,n) -> ushort index in fragment-ordered LDS tile group
static __device__ __forceinline__ int b_fidx(int n, int k) {
    return (((n >> 4) * 32) + (n & 15) + (k & 16)) * 16 + (k & 15);
}

// ---------------------------------------------------------------------------
// Weight packing: fp32 -> bf16 fragment order
// ---------------------------------------------------------------------------
__global__ void pack_dense_frag(const float* __restrict__ w, unsigned short* __restrict__ o,
                                int M, int K, int Kpad) {
    int i = blockIdx.x * 256 + threadIdx.x;
    int KB = Kpad >> 5;
    if (i >= (M >> 4) * KB * 512) return;
    int e    = i & 15;
    int lane = (i >> 4) & 31;
    int blk  = i >> 9;
    int mt = blk / KB, kb = blk - mt * KB;
    int m = mt * 16 + (lane & 15);
    int k = kb * 32 + a_koff(e, lane);
    float v = (k < K) ? w[m * K + k] : 0.f;
    o[i] = bfbits(v);
}

// conv3x3 weights [O][Cin][3][3], logical K = tap*Cin + cin, padded O
__global__ void pack_conv_frag(const float* __restrict__ w, unsigned short* __restrict__ o,
                               int O, int Opad, int Cin) {
    int K = 9 * Cin, KB = K >> 5;
    int i = blockIdx.x * 256 + threadIdx.x;
    if (i >= (Opad >> 4) * KB * 512) return;
    int e    = i & 15;
    int lane = (i >> 4) & 31;
    int blk  = i >> 9;
    int mt = blk / KB, kb = blk - mt * KB;
    int m = mt * 16 + (lane & 15);
    int k = kb * 32 + a_koff(e, lane);
    int tap = k / Cin, cin = k - tap * Cin;
    float v = (m < O) ? w[(m * Cin + cin) * 9 + tap] : 0.f;
    o[i] = bfbits(v);
}

// ---------------------------------------------------------------------------
// Generic WMMA GEMM: Out = Wbf16[64,Kpad] x In (+bias), N = 64 pixels/block
//   IN_LAYOUT  0: In[k*PIX + pix]   1: In[pix*64 + k]  (layout 1 => Kpad==64)
//   OUT_LAYOUT 0: Out[m*PIX + pix]  1: Out[pix*64 + m]
// grid.x = PIX/64, block = 128 (4 waves = 4 M-tiles)
// ---------------------------------------------------------------------------
template<int IN_LAYOUT, int OUT_LAYOUT>
__global__ __launch_bounds__(128)
void gemm_wmma(const v16us* __restrict__ Wf, const float* __restrict__ In,
               float* __restrict__ Out, const float* __restrict__ bias,
               int Kreal, int Kpad) {
    __shared__ v16us BtV[2][NT2 * 32];
    const int tid  = threadIdx.x;
    const int lane = tid & 31;
    const int mt   = tid >> 5;
    const int pix0 = blockIdx.x * (16 * NT2);
    const int KB   = Kpad >> 5;

    auto stage = [&](int kb, int buf) {
        unsigned short* Bt = (unsigned short*)BtV[buf];
        if (IN_LAYOUT == 0) {
            // channel-major: float4 along n (aligned), branchless K-tail select
            for (int idx = tid; idx < 512; idx += 128) {
                int n4 = idx & 15, k = idx >> 4;
                int kg  = kb * 32 + k;
                int kgc = min(kg, Kreal - 1);
                float4 f = *(const float4*)(In + (size_t)kgc * PIX + pix0 + n4 * 4);
                bool ok = (kg < Kreal);
                float vv[4] = {f.x, f.y, f.z, f.w};
#pragma unroll
                for (int j = 0; j < 4; ++j) {
                    int n = n4 * 4 + j;
                    Bt[b_fidx(n, k)] = bfbits(ok ? vv[j] : 0.f);
                }
            }
        } else {
            // pixel-major: float4 along k, 4 packed bf16 -> one b64 LDS store
            for (int idx = tid; idx < 512; idx += 128) {
                int n = idx & 63, k = (idx >> 6) * 4;
                float4 f = *(const float4*)(In + (size_t)(pix0 + n) * 64 + kb * 32 + k);
                unsigned u0 = ((unsigned)bfbits(f.y) << 16) | bfbits(f.x);
                unsigned u1 = ((unsigned)bfbits(f.w) << 16) | bfbits(f.z);
                *(uint2*)&Bt[b_fidx(n, k)] = make_uint2(u0, u1);
            }
        }
    };

    v8f acc[NT2] = {};
    stage(0, 0);
    __syncthreads();
    for (int kb = 0; kb < KB; ++kb) {
        const int cur = kb & 1;
        if (kb + 1 < KB) stage(kb + 1, cur ^ 1);
        v16us au = Wf[(size_t)(mt * KB + kb) * 32 + lane];
        v16bf a  = __builtin_bit_cast(v16bf, au);
        v16bf b[NT2];
#pragma unroll
        for (int nt = 0; nt < NT2; ++nt)          // issue all B ds_loads first
            b[nt] = __builtin_bit_cast(v16bf, BtV[cur][nt * 32 + lane]);
#pragma unroll
        for (int nt = 0; nt < NT2; ++nt)          // then drain the WMMA chain
            acc[nt] = __builtin_amdgcn_wmma_f32_16x16x32_bf16(false, a, false, b[nt],
                                                              (short)0, acc[nt], false, false);
        __syncthreads();
    }
#pragma unroll
    for (int nt = 0; nt < NT2; ++nt) {
#pragma unroll
        for (int r = 0; r < 8; ++r) {
            int m = mt * 16 + r + ((lane & 16) ? 8 : 0);
            int n = pix0 + nt * 16 + (lane & 15);
            float v = acc[nt][r] + (bias ? bias[m] : 0.f);
            if (OUT_LAYOUT == 0) Out[(size_t)m * PIX + n] = v;
            else                 Out[(size_t)n * 64 + m]  = v;
        }
    }
}

// ---------------------------------------------------------------------------
// Implicit-GEMM dilated conv3x3 (Cin=64, K=576).  One tap per 32-wide K-block
// so tap/row math is scalar; staging is branchless (clamped address + select).
// grid.x = PIX/64 (64-pixel row segments), block = 32*Mtiles.
// ---------------------------------------------------------------------------
__global__ __launch_bounds__(128)
void conv3x3_wmma(const v16us* __restrict__ Wf, const float* __restrict__ In,
                  float* __restrict__ Out, const float* __restrict__ bias,
                  int dil, int Cout) {
    __shared__ v16us BtV[2][NT2 * 32];
    const int KB   = 18;                     // 576/32
    const int tid  = threadIdx.x;
    const int lane = tid & 31;
    const int mt   = tid >> 5;
    const int t    = blockIdx.x;
    const int py   = t >> 2;                 // 4 x 64-pixel tiles per row
    const int px0  = (t & 3) << 6;
    const int nthr = blockDim.x;

    auto stage = [&](int kb, int buf) {
        unsigned short* Bt = (unsigned short*)BtV[buf];
        const int tap  = kb >> 1;            // 2 K-blocks per tap
        const int cin0 = (kb & 1) << 5;
        const int di = tap / 3 - 1, dj = tap % 3 - 1;
        const int yy  = py + di * dil;
        const int yyc = min(max(yy, 0), L - 1);
        const bool rowok = (yy >= 0) && (yy < L);
        const int djd = dj * dil;
        for (int idx = tid; idx < 32 * 16 * NT2; idx += nthr) {
            int n = idx & 63, k = idx >> 6;
            int xx  = px0 + n + djd;
            int xxc = min(max(xx, 0), L - 1);
            float v = In[(size_t)(cin0 + k) * PIX + yyc * L + xxc];
            v = (rowok && xx >= 0 && xx < L) ? v : 0.f;
            Bt[b_fidx(n, k)] = bfbits(v);
        }
    };

    v8f acc[NT2] = {};
    stage(0, 0);
    __syncthreads();
    for (int kb = 0; kb < KB; ++kb) {
        const int cur = kb & 1;
        if (kb + 1 < KB) stage(kb + 1, cur ^ 1);
        v16us au = Wf[(size_t)(mt * KB + kb) * 32 + lane];
        v16bf a  = __builtin_bit_cast(v16bf, au);
        v16bf b[NT2];
#pragma unroll
        for (int nt = 0; nt < NT2; ++nt)
            b[nt] = __builtin_bit_cast(v16bf, BtV[cur][nt * 32 + lane]);
#pragma unroll
        for (int nt = 0; nt < NT2; ++nt)
            acc[nt] = __builtin_amdgcn_wmma_f32_16x16x32_bf16(false, a, false, b[nt],
                                                              (short)0, acc[nt], false, false);
        __syncthreads();
    }
#pragma unroll
    for (int nt = 0; nt < NT2; ++nt) {
#pragma unroll
        for (int r = 0; r < 8; ++r) {
            int m = mt * 16 + r + ((lane & 16) ? 8 : 0);
            if (m < Cout) {
                float v = acc[nt][r] + (bias ? bias[m] : 0.f);
                Out[(size_t)m * PIX + py * L + px0 + nt * 16 + (lane & 15)] = v;
            }
        }
    }
}

// ---------------------------------------------------------------------------
// Instance-norm statistics: one block per channel -> stats[c] = {mean, rstd}
// ---------------------------------------------------------------------------
__global__ __launch_bounds__(256)
void inorm_stats(const float* __restrict__ X, float* __restrict__ stats) {
    __shared__ float sh[256], sh2[256];
    const int c = blockIdx.x;
    const float* x = X + (size_t)c * PIX;
    float s = 0.f, s2 = 0.f;
    for (int i = threadIdx.x; i < PIX; i += 256) {
        float v = x[i];
        s += v; s2 += v * v;
    }
    sh[threadIdx.x] = s; sh2[threadIdx.x] = s2;
    __syncthreads();
    for (int o = 128; o > 0; o >>= 1) {
        if (threadIdx.x < o) { sh[threadIdx.x] += sh[threadIdx.x + o];
                               sh2[threadIdx.x] += sh2[threadIdx.x + o]; }
        __syncthreads();
    }
    if (threadIdx.x == 0) {
        float mean = sh[0] * (1.f / PIX);
        float var  = sh2[0] * (1.f / PIX) - mean * mean;
        stats[2 * c]     = mean;
        stats[2 * c + 1] = rsqrtf(var + 1e-5f);
    }
}

__global__ __launch_bounds__(256)
void norm_relu(const float* __restrict__ in, const float* __restrict__ stats,
               float* __restrict__ out) {
    size_t i = (size_t)blockIdx.x * 256 + threadIdx.x;
    if (i >= (size_t)CH * PIX) return;
    int c = (int)(i >> 16);
    float v = (in[i] - stats[2 * c]) * stats[2 * c + 1];
    out[i] = fmaxf(v, 0.f);
}

__global__ __launch_bounds__(256)
void norm_add_relu(const float* __restrict__ t, const float* __restrict__ stats,
                   float* __restrict__ h) {
    size_t i = (size_t)blockIdx.x * 256 + threadIdx.x;
    if (i >= (size_t)CH * PIX) return;
    int c = (int)(i >> 16);
    float v = (t[i] - stats[2 * c]) * stats[2 * c + 1] + h[i];
    h[i] = fmaxf(v, 0.f);
}

// ---------------------------------------------------------------------------
// Regional 3x3 attention, 4 heads x 16 ch.  Q,K,V,O are [PIX][64] fp32.
// Matches reference: OOB neighbors contribute logit 0 and value 0.
// ---------------------------------------------------------------------------
__global__ __launch_bounds__(256)
void regional_attn(const float* __restrict__ Q, const float* __restrict__ Kb,
                   const float* __restrict__ V, float* __restrict__ O) {
    int gid = blockIdx.x * 256 + threadIdx.x;     // 4 * PIX threads
    int hd = gid >> 16;
    int p  = gid & (PIX - 1);
    int y = p >> 8, x = p & 255;

    float qr[16];
    const float* q = Q + (size_t)p * 64 + hd * 16;
#pragma unroll
    for (int c = 0; c < 16; ++c) qr[c] = q[c];

    float logit[9];
    float mx = -1e30f;
#pragma unroll
    for (int n = 0; n < 9; ++n) {
        int di = n / 3 - 1, dj = n % 3 - 1;
        int yy = y + di, xx = x + dj;
        float d = 0.f;
        if (yy >= 0 && yy < L && xx >= 0 && xx < L) {
            const float* k = Kb + ((size_t)(yy * L + xx)) * 64 + hd * 16;
#pragma unroll
            for (int c = 0; c < 16; ++c) d += qr[c] * k[c];
            d *= 0.125f;                       // 1/sqrt(64)
        }
        logit[n] = d;
        mx = fmaxf(mx, d);
    }
    float w[9], se = 0.f;
#pragma unroll
    for (int n = 0; n < 9; ++n) { w[n] = __expf(logit[n] - mx); se += w[n]; }
    float inv = 1.f / se;

    float o[16];
#pragma unroll
    for (int c = 0; c < 16; ++c) o[c] = 0.f;
#pragma unroll
    for (int n = 0; n < 9; ++n) {
        int di = n / 3 - 1, dj = n % 3 - 1;
        int yy = y + di, xx = x + dj;
        if (yy >= 0 && yy < L && xx >= 0 && xx < L) {
            const float* v = V + ((size_t)(yy * L + xx)) * 64 + hd * 16;
            float wn = w[n] * inv;
#pragma unroll
            for (int c = 0; c < 16; ++c) o[c] += wn * v[c];
        }
    }
    float* op = O + (size_t)p * 64 + hd * 16;
#pragma unroll
    for (int c = 0; c < 16; ++c) op[c] = o[c];
}

// ---------------------------------------------------------------------------
// Host driver
// ---------------------------------------------------------------------------
static inline size_t align256(size_t x) { return (x + 255) & ~(size_t)255; }

extern "C" void kernel_launch(void* const* d_in, const int* in_sizes, int n_in,
                              void* d_out, int out_size, void* d_ws, size_t ws_size,
                              hipStream_t stream) {
    const float* x    = (const float*)d_in[0];
    const float* c1_w = (const float*)d_in[1];
    const float* c1_b = (const float*)d_in[2];
    const float* bw1[3] = {(const float*)d_in[3], (const float*)d_in[7],  (const float*)d_in[11]};
    const float* bb1[3] = {(const float*)d_in[4], (const float*)d_in[8],  (const float*)d_in[12]};
    const float* bw2[3] = {(const float*)d_in[5], (const float*)d_in[9],  (const float*)d_in[13]};
    const float* bb2[3] = {(const float*)d_in[6], (const float*)d_in[10], (const float*)d_in[14]};
    const float* f1_w = (const float*)d_in[15]; const float* f1_b = (const float*)d_in[16];
    const float* wq = (const float*)d_in[17];
    const float* wk = (const float*)d_in[18];
    const float* wv = (const float*)d_in[19];
    const float* wo = (const float*)d_in[20];
    const float* bo = (const float*)d_in[21];
    const float* f2_w = (const float*)d_in[22]; const float* f2_b = (const float*)d_in[23];

    char* ws = (char*)d_ws;
    const size_t FM = (size_t)CH * PIX * sizeof(float);      // 16.78 MB
    float* h  = (float*)(ws);
    float* t0 = (float*)(ws + FM);
    float* t1 = (float*)(ws + 2 * FM);
    float* t2 = (float*)(ws + 3 * FM);
    float* stats = (float*)(ws + 4 * FM);
    char* wp = ws + 4 * FM + 4096;

    unsigned short* wc1p = (unsigned short*)wp; wp += align256((size_t)64 * 448 * 2);
    unsigned short* wcv[7];
    for (int i = 0; i < 7; ++i) { wcv[i] = (unsigned short*)wp; wp += align256((size_t)64 * 576 * 2); }
    unsigned short* wqp = (unsigned short*)wp; wp += align256((size_t)64 * 64 * 2);
    unsigned short* wkp = (unsigned short*)wp; wp += align256((size_t)64 * 64 * 2);
    unsigned short* wvp = (unsigned short*)wp; wp += align256((size_t)64 * 64 * 2);
    unsigned short* wop = (unsigned short*)wp; wp += align256((size_t)64 * 64 * 2);
    unsigned short* wf2p = (unsigned short*)wp; wp += align256((size_t)16 * 576 * 2);

    const int NB = PIX / (16 * NT2);         // 1024 blocks (64 pixels each)
    const int EG = (CH * PIX) / 256;         // 16384 elementwise blocks

    // ---- pack weights into bf16 fragment order ----
    pack_dense_frag<<<(64 * 448 + 255) / 256, 256, 0, stream>>>(c1_w, wc1p, 64, 441, 448);
    const float* cw[7] = {bw1[0], bw2[0], bw1[1], bw2[1], bw1[2], bw2[2], f1_w};
    for (int i = 0; i < 7; ++i)
        pack_conv_frag<<<(64 * 576 + 255) / 256, 256, 0, stream>>>(cw[i], wcv[i], 64, 64, 64);
    pack_dense_frag<<<(64 * 64 + 255) / 256, 256, 0, stream>>>(wq, wqp, 64, 64, 64);
    pack_dense_frag<<<(64 * 64 + 255) / 256, 256, 0, stream>>>(wk, wkp, 64, 64, 64);
    pack_dense_frag<<<(64 * 64 + 255) / 256, 256, 0, stream>>>(wv, wvp, 64, 64, 64);
    pack_dense_frag<<<(64 * 64 + 255) / 256, 256, 0, stream>>>(wo, wop, 64, 64, 64);
    pack_conv_frag<<<(16 * 576 + 255) / 256, 256, 0, stream>>>(f2_w, wf2p, 10, 16, 64);

    // ---- conv1x1 (441->64) + inorm + relu ----
    gemm_wmma<0,0><<<NB, 128, 0, stream>>>((const v16us*)wc1p, x, t0, c1_b, 441, 448);
    inorm_stats<<<64, 256, 0, stream>>>(t0, stats);
    norm_relu<<<EG, 256, 0, stream>>>(t0, stats, h);

    // ---- 3 dilated residual blocks ----
    const int dils[3] = {1, 2, 4};
    for (int i = 0; i < 3; ++i) {
        conv3x3_wmma<<<NB, 128, 0, stream>>>((const v16us*)wcv[2 * i], h, t0, bb1[i], dils[i], 64);
        inorm_stats<<<64, 256, 0, stream>>>(t0, stats);
        norm_relu<<<EG, 256, 0, stream>>>(t0, stats, t1);
        conv3x3_wmma<<<NB, 128, 0, stream>>>((const v16us*)wcv[2 * i + 1], t1, t2, bb2[i], dils[i], 64);
        inorm_stats<<<64, 256, 0, stream>>>(t2, stats);
        norm_add_relu<<<EG, 256, 0, stream>>>(t2, stats, h);
    }

    // ---- f1 conv + inorm + relu ----
    conv3x3_wmma<<<NB, 128, 0, stream>>>((const v16us*)wcv[6], h, t0, f1_b, 1, 64);
    inorm_stats<<<64, 256, 0, stream>>>(t0, stats);
    norm_relu<<<EG, 256, 0, stream>>>(t0, stats, h);

    // ---- regional attention ----
    gemm_wmma<0,1><<<NB, 128, 0, stream>>>((const v16us*)wqp, h, t0, nullptr, 64, 64);
    gemm_wmma<0,1><<<NB, 128, 0, stream>>>((const v16us*)wkp, h, t1, nullptr, 64, 64);
    gemm_wmma<0,1><<<NB, 128, 0, stream>>>((const v16us*)wvp, h, t2, nullptr, 64, 64);
    regional_attn<<<(4 * PIX) / 256, 256, 0, stream>>>(t0, t1, t2, h);
    gemm_wmma<1,0><<<NB, 128, 0, stream>>>((const v16us*)wop, h, t0, bo, 64, 64);

    // ---- f2 conv3x3 64->10 into d_out ----
    conv3x3_wmma<<<NB, 32, 0, stream>>>((const v16us*)wf2p, t0, (float*)d_out, f2_b, 1, 10);
}
